// LinearInceptionAttention_11484742550045
// MI455X (gfx1250) — matrix operand order repaired and checked by reading, not compiled, not run on hardware
//
#include <hip/hip_runtime.h>
#include <hip/hip_bf16.h>

typedef __bf16 bf16;
typedef __attribute__((ext_vector_type(16))) __bf16 v16bf;
typedef __attribute__((ext_vector_type(8)))  float  v8f;

#define EPSF 1e-6f

// ---------------- WMMA fragment helpers (contiguous k order: k = 16*(lane>>4) + element) ----------------

// A-matrix 16x32 bf16 from row-major f32 X[ldx]: lane l -> row m0+(l&15);
// VGPR v holds K = (l>>4)*16 + v*2 (+1). One 64B contiguous f32 chunk per lane.
__device__ __forceinline__ v16bf load_a_frag(const float* __restrict__ X, int ldx,
                                             int m0, int k0, int lane) {
  const float* xr = X + (size_t)(m0 + (lane & 15)) * ldx + k0;
  const int half = lane >> 4;
  v16bf a;
#pragma unroll
  for (int v = 0; v < 8; ++v) {
    int kb = (half << 4) + (v << 1);
    a[2 * v]     = (bf16)xr[kb];
    a[2 * v + 1] = (bf16)xr[kb + 1];
  }
  return a;
}

// Pre-packed B fragment: PB[(kt*16 + nt)*32 + lane][16] bf16, one 32B load per lane.
__device__ __forceinline__ v16bf load_b_packed(const bf16* __restrict__ PB,
                                               int kt, int nt, int lane) {
  return *(const v16bf*)(PB + (((size_t)(kt * 16 + nt) * 32 + lane) << 4));
}

__device__ __forceinline__ float elup1(float x) {   // elu(x) + 1 + eps
  return x > 0.0f ? (x + 1.0f + EPSF) : (__expf(x) + EPSF);
}

// ---------------- Kernel 0: pack f32 weight [Kdim x 256] into bf16 B-fragment tiles ----------
// B 32x16 layout: lane l -> col nt*16+(l&15); K = kt*32 + (l>>4)*16 + 2v (+1).
__global__ void __launch_bounds__(256)
pack_b_kernel(const float* __restrict__ W, int Kdim, bf16* __restrict__ PB) {
  int t = blockIdx.x * blockDim.x + threadIdx.x;       // (Kdim/32)*16 tiles * 32 lanes
  int lane = t & 31;
  int tile = t >> 5;
  int ktiles = Kdim >> 5;
  if (tile >= ktiles * 16) return;
  int nt = tile & 15, kt = tile >> 4;
  int col = nt * 16 + (lane & 15);
  int kb0 = kt * 32 + ((lane >> 4) << 4);
  bf16* dst = PB + ((size_t)t << 4);                   // 32B per thread, coalesced
#pragma unroll
  for (int v = 0; v < 8; ++v) {
    dst[2 * v]     = (bf16)W[(size_t)(kb0 + 2 * v) * 256 + col];
    dst[2 * v + 1] = (bf16)W[(size_t)(kb0 + 2 * v + 1) * 256 + col];
  }
}

// ---------------- Kernel 1: fused QKV projection + ELU feature map ----------------
__global__ void __launch_bounds__(256)
qkv_proj_kernel(const float* __restrict__ x,
                const bf16* __restrict__ pwq, const bf16* __restrict__ pwk,
                const bf16* __restrict__ pwv,
                float* __restrict__ Q, float* __restrict__ K, float* __restrict__ V) {
  const int lane = threadIdx.x & 31;
  const int gw = blockIdx.x * 8 + (threadIdx.x >> 5);   // 4096 wave-tiles
  const int m0 = (gw >> 4) * 16;
  const int nt = gw & 15;
  const int n0 = nt * 16;

  v8f cq = {}, ck = {}, cv = {};
#pragma unroll
  for (int kt = 0; kt < 8; ++kt) {
    v16bf a  = load_a_frag(x, 256, m0, kt * 32, lane);
    v16bf bq = load_b_packed(pwq, kt, nt, lane);
    v16bf bk = load_b_packed(pwk, kt, nt, lane);
    v16bf bv = load_b_packed(pwv, kt, nt, lane);
    cq = __builtin_amdgcn_wmma_f32_16x16x32_bf16(false, a, false, bq, (short)0, cq, false, false);
    ck = __builtin_amdgcn_wmma_f32_16x16x32_bf16(false, a, false, bk, (short)0, ck, false, false);
    cv = __builtin_amdgcn_wmma_f32_16x16x32_bf16(false, a, false, bv, (short)0, cv, false, false);
  }
  const int col  = n0 + (lane & 15);
  const int head = col >> 5, ch = col & 31;
  const int mb   = m0 + 8 * (lane >> 4);
#pragma unroll
  for (int v = 0; v < 8; ++v) {
    size_t idx = ((size_t)head * 4096 + (mb + v)) * 32 + ch;
    Q[idx] = elup1(cq[v]);
    K[idx] = elup1(ck[v]);
    V[idx] = cv[v];
  }
}

// ---------------- Kernel 2: K integral image (8,64,64,32) -> SK ----------------
__global__ void __launch_bounds__(256)
cumsum_k_row(const float* __restrict__ K, float* __restrict__ SK) {
  int t = blockIdx.x * blockDim.x + threadIdx.x;      // 8*64*32 = 16384
  int ch = t & 31, i = (t >> 5) & 63, head = t >> 11;
  const float* src = K  + ((size_t)head * 4096 + i * 64) * 32 + ch;
  float*       dst = SK + ((size_t)head * 4096 + i * 64) * 32 + ch;
  float acc = 0.0f;
  for (int j = 0; j < 64; ++j) { acc += src[j * 32]; dst[j * 32] = acc; }
}

__global__ void __launch_bounds__(256)
cumsum_k_col(float* __restrict__ SK) {
  int t = blockIdx.x * blockDim.x + threadIdx.x;
  int ch = t & 31, j = (t >> 5) & 63, head = t >> 11;
  float* p = SK + ((size_t)head * 4096 + j) * 32 + ch;
  float acc = 0.0f;
  for (int i = 0; i < 64; ++i) {
    float v = p[(size_t)i * 64 * 32]; acc += v; p[(size_t)i * 64 * 32] = acc;
  }
}

// ---------------- Kernel 3: KV outer-product integral image (8,64,64,32,32) ----------------
__global__ void __launch_bounds__(256)
kv_row_kernel(const float* __restrict__ K, const float* __restrict__ V,
              float* __restrict__ SKV) {
  __shared__ float lk[32], lv[32];
  const int head = blockIdx.x >> 6;
  const int i    = blockIdx.x & 63;
  const int t    = threadIdx.x;
  const float* kp = K + ((size_t)head * 4096 + i * 64) * 32;
  const float* vp = V + ((size_t)head * 4096 + i * 64) * 32;
  float* out = SKV + ((size_t)head * 64 + i) * 64 * 1024;
  float acc[4] = {0.f, 0.f, 0.f, 0.f};
  for (int j = 0; j < 64; ++j) {
    __syncthreads();
    if (t < 32)       lk[t]      = kp[j * 32 + t];
    else if (t < 64)  lv[t - 32] = vp[j * 32 + (t - 32)];
    __syncthreads();
#pragma unroll
    for (int r = 0; r < 4; ++r) {
      int idx = t + 256 * r;                       // d = idx>>5, e = idx&31
      acc[r] += lk[idx >> 5] * lv[idx & 31];
      out[(size_t)j * 1024 + idx] = acc[r];
    }
  }
}

__global__ void __launch_bounds__(256)
kv_col_kernel(float* __restrict__ SKV) {
  const int head = blockIdx.x >> 6;
  const int j    = blockIdx.x & 63;
  const int t    = threadIdx.x;
  float* base = SKV + ((size_t)head * 64 * 64 + j) * 1024;
  float acc[4] = {0.f, 0.f, 0.f, 0.f};
  for (int i = 0; i < 64; ++i) {
    float* p = base + (size_t)i * 64 * 1024;
#pragma unroll
    for (int r = 0; r < 4; ++r) {
      int idx = t + 256 * r;
      float v = p[idx]; acc[r] += v; p[idx] = acc[r];
    }
  }
}

// ---------------- Kernel 4: windowed box retrieval + linear attention ----------------
// One wave per (window, head, position). Corner matrices (4KB each) streamed as
// contiguous float4 per lane (full 128B transactions); Ikv is L2-resident (128MiB < 192MB).
__global__ void __launch_bounds__(256)
attn_box_kernel(const float* __restrict__ Q, const float* __restrict__ SK,
                const float* __restrict__ SKV, float* __restrict__ O) {
  const int lane = threadIdx.x & 31;
  const int g    = blockIdx.x * 8 + (threadIdx.x >> 5);   // 3*8*4096 waves
  const int win  = g >> 15;
  const int hp   = g & 32767;
  const int head = hp >> 12;
  const int p    = hp & 4095;
  const int i = p >> 6, j = p & 63;
  const int r = 32 >> win;                                // radii 32,16,8
  const int x2 = min(i + r, 63), y2 = min(j + r, 63);
  const int a1 = i - r - 1, b1 = j - r - 1;               // <0 -> zero border
  const bool u01 = (a1 >= 0), u10 = (b1 >= 0);

  const float* Hb  = SKV + (size_t)head * 64 * 64 * 1024;
  const float* P11 = Hb + ((size_t)x2 * 64 + y2) * 1024;
  const float* P01 = u01 ? Hb + ((size_t)a1 * 64 + y2) * 1024 : P11;
  const float* P10 = u10 ? Hb + ((size_t)x2 * 64 + b1) * 1024 : P11;
  const float* P00 = (u01 && u10) ? Hb + ((size_t)a1 * 64 + b1) * 1024 : P11;
  __builtin_prefetch(P11, 0, 1);
  if (u01) __builtin_prefetch(P01, 0, 1);
  if (u10) __builtin_prefetch(P10, 0, 1);
  if (u01 && u10) __builtin_prefetch(P00, 0, 1);

  const float qv = Q[((size_t)head * 4096 + p) * 32 + lane];   // lane = channel d

  // denominator: q . box(SK)
  const float* Sh = SK + (size_t)head * 4096 * 32;
  float kb = Sh[(x2 * 64 + y2) * 32 + lane];
  if (u01)        kb -= Sh[(a1 * 64 + y2) * 32 + lane];
  if (u10)        kb -= Sh[(x2 * 64 + b1) * 32 + lane];
  if (u01 && u10) kb += Sh[(a1 * 64 + b1) * 32 + lane];
  float den = qv * kb;
#pragma unroll
  for (int s = 16; s > 0; s >>= 1) den += __shfl_xor(den, s, 32);
  den += EPSF;

  // numerator: q^T . box(SKV). Lane covers e = (lane&7)*4..+3, rows d = it*4 + lane/8.
  float n0 = 0.f, n1 = 0.f, n2 = 0.f, n3 = 0.f;
#pragma unroll
  for (int it = 0; it < 8; ++it) {
    const int base = it * 128 + lane * 4;
    const int d    = base >> 5;
    float4 s = *(const float4*)(P11 + base);
    if (u01) { float4 t = *(const float4*)(P01 + base); s.x -= t.x; s.y -= t.y; s.z -= t.z; s.w -= t.w; }
    if (u10) { float4 t = *(const float4*)(P10 + base); s.x -= t.x; s.y -= t.y; s.z -= t.z; s.w -= t.w; }
    if (u01 && u10) { float4 t = *(const float4*)(P00 + base); s.x += t.x; s.y += t.y; s.z += t.z; s.w += t.w; }
    const float qd = __shfl(qv, d, 32);
    n0 = fmaf(qd, s.x, n0); n1 = fmaf(qd, s.y, n1);
    n2 = fmaf(qd, s.z, n2); n3 = fmaf(qd, s.w, n3);
  }
  // combine the 4 d-row partials per e-group (lanes l, l+8, l+16, l+24)
  n0 += __shfl_xor(n0, 8, 32);  n0 += __shfl_xor(n0, 16, 32);
  n1 += __shfl_xor(n1, 8, 32);  n1 += __shfl_xor(n1, 16, 32);
  n2 += __shfl_xor(n2, 8, 32);  n2 += __shfl_xor(n2, 16, 32);
  n3 += __shfl_xor(n3, 8, 32);  n3 += __shfl_xor(n3, 16, 32);

  const float inv = 1.0f / den;
  if (lane < 8) {
    float4 o4 = make_float4(n0 * inv, n1 * inv, n2 * inv, n3 * inv);
    *(float4*)(O + (size_t)p * 768 + win * 256 + head * 32 + lane * 4) = o4;
  }
}

// ---------------- Kernel 5: output projection O[4096,768] @ wo[768,256] + bo ----------------
__global__ void __launch_bounds__(256)
out_proj_kernel(const float* __restrict__ O, const bf16* __restrict__ pwo,
                const float* __restrict__ BO, float* __restrict__ out) {
  const int lane = threadIdx.x & 31;
  const int gw = blockIdx.x * 8 + (threadIdx.x >> 5);   // 4096 wave-tiles
  const int m0 = (gw >> 4) * 16;
  const int nt = gw & 15;
  v8f c = {};
#pragma unroll
  for (int kt = 0; kt < 24; ++kt) {
    v16bf a = load_a_frag(O, 768, m0, kt * 32, lane);
    v16bf b = load_b_packed(pwo, kt, nt, lane);
    c = __builtin_amdgcn_wmma_f32_16x16x32_bf16(false, a, false, b, (short)0, c, false, false);
  }
  const int col = nt * 16 + (lane & 15);
  const float bias = BO[col];
  const int mb = m0 + 8 * (lane >> 4);
#pragma unroll
  for (int v = 0; v < 8; ++v)
    out[(size_t)(mb + v) * 256 + col] = c[v] + bias;
}

// ---------------- launch ----------------
extern "C" void kernel_launch(void* const* d_in, const int* in_sizes, int n_in,
                              void* d_out, int out_size, void* d_ws, size_t ws_size,
                              hipStream_t stream) {
  const float* x  = (const float*)d_in[0];
  const float* wq = (const float*)d_in[1];
  const float* wk = (const float*)d_in[2];
  const float* wv = (const float*)d_in[3];
  const float* wo = (const float*)d_in[4];
  const float* bo = (const float*)d_in[5];
  float* out = (float*)d_out;

  char* ws = (char*)d_ws;
  float* Q   = (float*)(ws);                               //  4 MiB (8,4096,32)
  float* K   = (float*)(ws + ( 4ull << 20));               //  4 MiB
  float* V   = (float*)(ws + ( 8ull << 20));               //  4 MiB
  float* SK  = (float*)(ws + (12ull << 20));               //  4 MiB
  float* O   = (float*)(ws + (16ull << 20));               // 12 MiB (4096,768)
  bf16*  PWQ = (bf16*)(ws + (28ull << 20));                // 128 KiB packed B tiles
  bf16*  PWK = (bf16*)(ws + (28ull << 20) + (128ull << 10));
  bf16*  PWV = (bf16*)(ws + (28ull << 20) + (256ull << 10));
  bf16*  PWO = (bf16*)(ws + (28ull << 20) + (384ull << 10)); // 384 KiB
  float* SKV = (float*)(ws + (32ull << 20));               // 128 MiB (fits in L2)

  pack_b_kernel<<<16, 256, 0, stream>>>(wq, 256, PWQ);
  pack_b_kernel<<<16, 256, 0, stream>>>(wk, 256, PWK);
  pack_b_kernel<<<16, 256, 0, stream>>>(wv, 256, PWV);
  pack_b_kernel<<<48, 256, 0, stream>>>(wo, 768, PWO);

  qkv_proj_kernel<<<512, 256, 0, stream>>>(x, PWQ, PWK, PWV, Q, K, V);
  cumsum_k_row<<<64, 256, 0, stream>>>(K, SK);
  cumsum_k_col<<<64, 256, 0, stream>>>(SK);
  kv_row_kernel<<<512, 256, 0, stream>>>(K, V, SKV);
  kv_col_kernel<<<512, 256, 0, stream>>>(SKV);
  attn_box_kernel<<<12288, 256, 0, stream>>>(Q, SK, SKV, O);
  out_proj_kernel<<<512, 256, 0, stream>>>(O, PWO, bo, out);
}
